// MemEffAttentionRope_30468497998145
// MI455X (gfx1250) — matrix operand order repaired and checked by reading, not compiled, not run on hardware
//
#include <hip/hip_runtime.h>
#include <math.h>

#define DIM   1024
#define HEADS 16
#define HDIM  64
#define BATCH 2
#define SEQ   2048

typedef __attribute__((ext_vector_type(16))) _Float16 v16h;
typedef __attribute__((ext_vector_type(8)))  float    v8f;
typedef __attribute__((ext_vector_type(4)))  unsigned v4u;
typedef __attribute__((ext_vector_type(4)))  int      v4i;
typedef __attribute__((ext_vector_type(8)))  int      v8i;

union FragH { unsigned u[8]; v16h h; };
union FragF { v8f v; float f[8]; };

__device__ __forceinline__ unsigned pack2h(float a, float b) {
  union { _Float16 h[2]; unsigned u; } p;
  p.h[0] = (_Float16)a; p.h[1] = (_Float16)b;
  return p.u;
}

template <short N>
__device__ __forceinline__ void wait_tensorcnt() {
#if __has_builtin(__builtin_amdgcn_s_wait_tensorcnt)
  __builtin_amdgcn_s_wait_tensorcnt(N);
#else
  asm volatile("s_wait_tensorcnt %0" :: "n"(N) : "memory");
#endif
}

// ---------------------------------------------------------------------------
// TDM: DMA a (tile_h x 32) f16 tile (row stride = row_stride elements) from
// global into LDS, contiguous. D# packed per CDNA5 ISA 08_async_tensor 8.3/8.4.
// Issue from one wave only (uniform branch); tracked by TENSORcnt.
// ---------------------------------------------------------------------------
__device__ __forceinline__ void tdm_load_tile_f16(
    void* lds_dst, const _Float16* gsrc, unsigned tile_h,
    unsigned row_stride, unsigned tensor_w, unsigned tensor_h) {
  const unsigned tile_w = 32u;
#if __has_builtin(__builtin_amdgcn_tensor_load_to_lds)
  unsigned long long ga = (unsigned long long)(size_t)gsrc;
  v4u g0;
  g0.x = 1u;                                               // count=1, user mode
  g0.y = (unsigned)(size_t)lds_dst;                        // lds_addr (bytes)
  g0.z = (unsigned)ga;                                     // global_addr[31:0]
  g0.w = ((unsigned)(ga >> 32) & 0x01FFFFFFu) | 0x80000000u; // [56:32] | type=2
  const unsigned long long s0 = row_stride;                // dim0 stride (elems)
  v8i g1;
  g1[0] = (int)(1u << 16);                                 // data_size = 2 bytes
  g1[1] = (int)((tensor_w & 0xFFFFu) << 16);               // tensor_dim0 lo16
  g1[2] = (int)((tensor_w >> 16) | ((tensor_h & 0xFFFFu) << 16));
  g1[3] = (int)((tensor_h >> 16) | (tile_w << 16));        // tile_dim0
  g1[4] = (int)(tile_h & 0xFFFFu);                         // tile_dim1, dim2=0
  g1[5] = (int)(unsigned)s0;                               // stride0 lo32
  g1[6] = (int)((unsigned)(s0 >> 32) & 0xFFFFu);           // stride0 hi16
  g1[7] = 0;
  v4i z4 = {0, 0, 0, 0};
#if __clang_major__ >= 23
  v8i z8 = {0, 0, 0, 0, 0, 0, 0, 0};
  __builtin_amdgcn_tensor_load_to_lds(g0, g1, z4, z4, z8, 0);
#else
  __builtin_amdgcn_tensor_load_to_lds(g0, g1, z4, z4, 0);
#endif
#else
  // fallback: wave-cooperative synchronous copy
  unsigned* ld = (unsigned*)lds_dst;
  const int lane = threadIdx.x & 31;
  const unsigned ppr = tile_w >> 1;
  for (unsigned r = 0; r < tile_h; ++r)
    for (unsigned p = lane; p < ppr; p += 32)
      ld[r * ppr + p] = *(const unsigned*)(gsrc + (size_t)r * row_stride + 2 * p);
#endif
}

// ---------------------------------------------------------------------------
// C[M,N] = A[M,K]*W[N,K]^T (+bias). A,W are f16 row-major K-contiguous, C f32.
// 256 threads = 8 waves, block tile 128x64, K-step 32, double-buffered TDM.
// Requires M%128==0, N%64==0, K%32==0.
// ---------------------------------------------------------------------------
__global__ __launch_bounds__(256) void gemm_h_wmma(
    const _Float16* __restrict__ A, const _Float16* __restrict__ W,
    const float* __restrict__ bias, float* __restrict__ C,
    int M, int N, int K) {
  __shared__ unsigned As[2][128 * 16];   // 128 rows x 32 f16 (16 uint pairs)
  __shared__ unsigned Bs[2][64 * 16];
  const int tid  = threadIdx.x;
  const int wave = tid >> 5, lane = tid & 31;
  const int bm = blockIdx.y * 128, bn = blockIdx.x * 64;

  FragF acc[4];
#pragma unroll
  for (int t = 0; t < 4; ++t)
#pragma unroll
    for (int r = 0; r < 8; ++r) acc[t].f[r] = 0.f;

  const int nk = K >> 5;
  if (wave == 0) {   // prologue: DMA tile 0 into buffer 0
    tdm_load_tile_f16(&As[0][0], A + (size_t)bm * K, 128, K, K, M);
    tdm_load_tile_f16(&Bs[0][0], W + (size_t)bn * K, 64, K, K, N);
  }

  for (int t = 0; t < nk; ++t) {
    const int cur = t & 1;
    if (wave == 0) {
      if (t + 1 < nk) {  // DMA next tile into other buffer, keep 1 pair in flight
        const size_t ko = (size_t)(t + 1) * 32;
        tdm_load_tile_f16(&As[cur ^ 1][0], A + (size_t)bm * K + ko, 128, K, K, M);
        tdm_load_tile_f16(&Bs[cur ^ 1][0], W + (size_t)bn * K + ko, 64, K, K, N);
        wait_tensorcnt<2>();   // oldest pair (tile t) complete
      } else {
        wait_tensorcnt<0>();
      }
    }
    __syncthreads();

    const unsigned* Au = &As[cur][0];
    const unsigned* Bu = &Bs[cur][0];
    // A fragment (16x32): lane<16 holds k 0..15, lane>=16 k 16..31
    FragH a;
    const int am = wave * 16 + (lane & 15);
    const int ab = (lane < 16) ? 0 : 8;
#pragma unroll
    for (int i = 0; i < 8; ++i)
      a.u[i] = Au[am * 16 + ab + i];
#pragma unroll
    for (int tt = 0; tt < 4; ++tt) {
      // B fragment (32x16): lane<16 col n k=0..15, lane>=16 same col k=16..31
      FragH b;
      const int bncol = tt * 16 + (lane & 15);
      const int bb = (lane < 16) ? 0 : 8;
#pragma unroll
      for (int i = 0; i < 8; ++i) b.u[i] = Bu[bncol * 16 + bb + i];
      acc[tt].v = __builtin_amdgcn_wmma_f32_16x16x32_f16(
          false, a.h, false, b.h, (short)0, acc[tt].v, false, false);
    }
    __syncthreads();  // all reads done before wave0 overwrites this buffer
  }

  const int mrow = bm + wave * 16 + ((lane < 16) ? 0 : 8);
#pragma unroll
  for (int t = 0; t < 4; ++t) {
    const int ncol = bn + t * 16 + (lane & 15);
    const float badd = bias ? bias[ncol] : 0.f;
#pragma unroll
    for (int r = 0; r < 8; ++r)
      C[(size_t)(mrow + r) * N + ncol] = acc[t].f[r] + badd;
  }
}

// ---------------------------------------------------------------------------
// f32 -> f16 conversion, 4 elements/thread (n must be a multiple of 4)
// ---------------------------------------------------------------------------
__global__ __launch_bounds__(256) void cvt_f32_f16(
    const float* __restrict__ s, _Float16* __restrict__ d, int n4) {
  const int i = blockIdx.x * 256 + threadIdx.x;
  if (i < n4) {
    const float4 v = ((const float4*)s)[i];
    ((uint2*)d)[i] = make_uint2(pack2h(v.x, v.y), pack2h(v.z, v.w));
  }
}

// ---------------------------------------------------------------------------
// Per-(b,n,h) LayerNorm over the 64-wide head dim for q and k, then RoPE.
// In place on qkv workspace [b][n][3][H][64]. 64 threads per block.
// ---------------------------------------------------------------------------
__global__ __launch_bounds__(64) void ln_rope_kernel(
    float* __restrict__ qkv, const int* __restrict__ pos,
    const float* __restrict__ qn_w, const float* __restrict__ qn_b,
    const float* __restrict__ kn_w, const float* __restrict__ kn_b, int N) {
  __shared__ float sq[64], sk[64], rq[64], rk[64];
  const int idx = blockIdx.x;
  const int h = idx % HEADS;
  const int n = (idx / HEADS) % N;
  const int b = idx / (HEADS * N);
  const int d = threadIdx.x;
  const size_t baseq = ((size_t)(b * N + n) * 3) * DIM + h * HDIM;
  const size_t basek = baseq + DIM;
  const float qv = qkv[baseq + d], kv = qkv[basek + d];
  sq[d] = qv; sk[d] = kv;
  __syncthreads();
  for (int s = 32; s > 0; s >>= 1) {
    if (d < s) { sq[d] += sq[d + s]; sk[d] += sk[d + s]; }
    __syncthreads();
  }
  const float mq = sq[0] * (1.f / 64.f), mk = sk[0] * (1.f / 64.f);
  __syncthreads();
  sq[d] = (qv - mq) * (qv - mq);
  sk[d] = (kv - mk) * (kv - mk);
  __syncthreads();
  for (int s = 32; s > 0; s >>= 1) {
    if (d < s) { sq[d] += sq[d + s]; sk[d] += sk[d + s]; }
    __syncthreads();
  }
  const float vq = sq[0] * (1.f / 64.f), vk = sk[0] * (1.f / 64.f);
  const float qn = (qv - mq) * rsqrtf(vq + 1e-5f) * qn_w[d] + qn_b[d];
  const float kn = (kv - mk) * rsqrtf(vk + 1e-5f) * kn_w[d] + kn_b[d];
  rq[d] = qn; rk[d] = kn;
  __syncthreads();
  const int j = d & 31;
  const float p = (float)pos[n];
  const float ang = p * __expf(-(float)j * (9.2103403719761836f / 32.f));
  const float c = cosf(ang), s = sinf(ang);
  float oq, ok;
  if (d < 32) { oq = rq[d] * c - rq[d + 32] * s; ok = rk[d] * c - rk[d + 32] * s; }
  else        { oq = rq[d - 32] * s + rq[d] * c; ok = rk[d - 32] * s + rk[d] * c; }
  qkv[baseq + d] = oq;
  qkv[basek + d] = ok;
}

// ---------------------------------------------------------------------------
// Flash attention, one wave per 16 q-rows of one (b,h). Streams keys/values in
// chunks of 32, online softmax via width-16 shuffles, WMMA for S and P*V.
// Output written in f16 (feeds the projection GEMM).
// ---------------------------------------------------------------------------
__global__ __launch_bounds__(32) void attn_flash_wmma(
    const float* __restrict__ qkv, _Float16* __restrict__ o, int N) {
  __shared__ float Pls[16 * 32];      // P tile staging: C-layout -> A-layout
  const int lane = threadIdx.x;
  const int qt = blockIdx.x;
  const int h = blockIdx.y % HEADS, b = blockIdx.y / HEADS;
  const size_t rstride = 3 * DIM;
  const float* qb = qkv + (size_t)b * N * rstride + h * HDIM;
  const float* kb = qb + DIM;
  const float* vb = qb + 2 * DIM;
  const float scale = 0.125f;  // 64^-0.5

  FragH qa[2];
  {
    const int m = qt * 16 + (lane & 15);
    const float* qrow = qb + (size_t)m * rstride;
    const int ab = (lane < 16) ? 0 : 8;
#pragma unroll
    for (int half = 0; half < 2; ++half)
#pragma unroll
      for (int i = 0; i < 8; ++i) {
        const int kp = ab + i;
        const int d = half * 32 + kp * 2;
        qa[half].u[i] = pack2h(qrow[d] * scale, qrow[d + 1] * scale);
      }
  }

  FragF acc[4];
  float mrow[8], lrow[8];
#pragma unroll
  for (int t = 0; t < 4; ++t)
#pragma unroll
    for (int r = 0; r < 8; ++r) acc[t].f[r] = 0.f;
#pragma unroll
  for (int r = 0; r < 8; ++r) { mrow[r] = -INFINITY; lrow[r] = 0.f; }

  for (int kc = 0; kc < N; kc += 32) {
    FragF st[2];
#pragma unroll
    for (int s = 0; s < 2; ++s) {
      FragF sf;
#pragma unroll
      for (int r = 0; r < 8; ++r) sf.f[r] = 0.f;
#pragma unroll
      for (int half = 0; half < 2; ++half) {
        FragH bk;  // B fragment: col = key, k = d
        const int col = kc + s * 16 + (lane & 15);
        const float* krow = kb + (size_t)col * rstride;
        const int d0 = half * 32 + ((lane < 16) ? 0 : 16);
#pragma unroll
        for (int i = 0; i < 8; ++i)
          bk.u[i] = pack2h(krow[d0 + 2 * i], krow[d0 + 2 * i + 1]);
        sf.v = __builtin_amdgcn_wmma_f32_16x16x32_f16(
            false, qa[half].h, false, bk.h, (short)0, sf.v, false, false);
      }
      st[s] = sf;
    }

    float fac[8];
#pragma unroll
    for (int r = 0; r < 8; ++r) {
      float v = fmaxf(st[0].f[r], st[1].f[r]);
#pragma unroll
      for (int off = 1; off < 16; off <<= 1) v = fmaxf(v, __shfl_xor(v, off, 16));
      const float mn = fmaxf(mrow[r], v);
      fac[r] = __expf(mrow[r] - mn);
      mrow[r] = mn;
      st[0].f[r] = __expf(st[0].f[r] - mn);
      st[1].f[r] = __expf(st[1].f[r] - mn);
      float rs = st[0].f[r] + st[1].f[r];
#pragma unroll
      for (int off = 1; off < 16; off <<= 1) rs += __shfl_xor(rs, off, 16);
      lrow[r] = lrow[r] * fac[r] + rs;
#pragma unroll
      for (int t = 0; t < 4; ++t) acc[t].f[r] *= fac[r];
    }

#pragma unroll
    for (int s = 0; s < 2; ++s)
#pragma unroll
      for (int r = 0; r < 8; ++r)
        Pls[(r + ((lane < 16) ? 0 : 8)) * 32 + s * 16 + (lane & 15)] = st[s].f[r];
    asm volatile("s_wait_dscnt 0" ::: "memory");  // same-wave LDS RAW ordering
    FragH pa;
    {
      const int m = lane & 15;
      const int ab = (lane < 16) ? 0 : 8;
#pragma unroll
      for (int i = 0; i < 8; ++i) {
        const int kp = ab + i;
        pa.u[i] = pack2h(Pls[m * 32 + kp * 2], Pls[m * 32 + kp * 2 + 1]);
      }
    }

#pragma unroll
    for (int t = 0; t < 4; ++t) {
      FragH bv;  // V B-fragment: col = d, k = key (row stride 3*DIM)
      const int dcol = t * 16 + (lane & 15);
      const int kk0 = (lane < 16) ? 0 : 16;
#pragma unroll
      for (int i = 0; i < 8; ++i)
        bv.u[i] = pack2h(vb[(size_t)(kc + kk0 + 2 * i) * rstride + dcol],
                         vb[(size_t)(kc + kk0 + 2 * i + 1) * rstride + dcol]);
      acc[t].v = __builtin_amdgcn_wmma_f32_16x16x32_f16(
          false, pa.h, false, bv.h, (short)0, acc[t].v, false, false);
    }
  }

  const int m0 = qt * 16 + ((lane < 16) ? 0 : 8);
#pragma unroll
  for (int t = 0; t < 4; ++t)
#pragma unroll
    for (int r = 0; r < 8; ++r)
      o[((size_t)(b * N) + m0 + r) * DIM + h * HDIM + t * 16 + (lane & 15)] =
          (_Float16)(acc[t].f[r] / lrow[r]);
}

// ---------------------------------------------------------------------------
extern "C" void kernel_launch(void* const* d_in, const int* in_sizes, int n_in,
                              void* d_out, int out_size, void* d_ws, size_t ws_size,
                              hipStream_t stream) {
  const float* x      = (const float*)d_in[0];
  const float* qkv_w  = (const float*)d_in[1];
  const float* qn_w   = (const float*)d_in[2];
  const float* qn_b   = (const float*)d_in[3];
  const float* kn_w   = (const float*)d_in[4];
  const float* kn_b   = (const float*)d_in[5];
  const float* proj_w = (const float*)d_in[6];
  const float* proj_b = (const float*)d_in[7];
  const int*   pos    = (const int*)d_in[8];
  float* out = (float*)d_out;

  const int B = BATCH, N = SEQ, C = DIM;
  const int M = B * N;  // 4096 token rows

  char* base = (char*)d_ws;
  float*     qkv = (float*)base;                                  // 48 MB f32
  _Float16*  xh  = (_Float16*)(base + (size_t)48 * 1024 * 1024);  //  8 MB
  _Float16*  wqh = (_Float16*)(base + (size_t)56 * 1024 * 1024);  //  6 MB
  _Float16*  wph = (_Float16*)(base + (size_t)62 * 1024 * 1024);  //  2 MB
  _Float16*  aoh = (_Float16*)(base + (size_t)64 * 1024 * 1024);  //  8 MB

  // 0) one-time f16 copies of GEMM operands (memory-bound, ~2us)
  cvt_f32_f16<<<dim3((M * C / 4 + 255) / 256), dim3(256), 0, stream>>>(x, xh, M * C / 4);
  cvt_f32_f16<<<dim3((3 * C * C / 4 + 255) / 256), dim3(256), 0, stream>>>(qkv_w, wqh, 3 * C * C / 4);
  cvt_f32_f16<<<dim3((C * C / 4 + 255) / 256), dim3(256), 0, stream>>>(proj_w, wph, C * C / 4);

  // 1) QKV projection: qkv = x @ qkv_w^T  (TDM-staged f16 WMMA)
  gemm_h_wmma<<<dim3(3 * C / 64, M / 128), dim3(256), 0, stream>>>(
      xh, wqh, nullptr, qkv, M, 3 * C, C);

  // 2) per-head LN on q,k + RoPE (in place, f32)
  ln_rope_kernel<<<dim3(B * N * HEADS), dim3(64), 0, stream>>>(
      qkv, pos, qn_w, qn_b, kn_w, kn_b, N);

  // 3) flash attention per (b,h), 16 q-rows per wave; f16 output
  attn_flash_wmma<<<dim3(N / 16, B * HEADS), dim3(32), 0, stream>>>(qkv, aoh, N);

  // 4) output projection: out = aoh @ proj_w^T + proj_b
  gemm_h_wmma<<<dim3(C / 64, M / 128), dim3(256), 0, stream>>>(
      aoh, wph, proj_b, out, M, C, C);
}
